// MultiheadAttention_10419590660779
// MI455X (gfx1250) — matrix operand order repaired and checked by reading, not compiled, not run on hardware
//
#include <hip/hip_runtime.h>
#include <math.h>

typedef __bf16 bf16;
typedef bf16 v16bf __attribute__((ext_vector_type(16)));
typedef bf16 v8bf  __attribute__((ext_vector_type(8)));
typedef float v8f  __attribute__((ext_vector_type(8)));

#define NSEQ   2048
#define NTOK   4096    /* B*N */
#define NDIM   1536
#define NHEAD  8
#define HD     64
#define HDK    512     /* H*DK */
#define NPOSP  4096    /* padded 2N-1 */

static __device__ __forceinline__ bf16 f2bf(float x) {
  union { float f; unsigned u; } v; v.f = x;
  unsigned r = v.u + 0x7FFFu + ((v.u >> 16) & 1u);
  unsigned short hs = (unsigned short)(r >> 16);
  bf16 o; __builtin_memcpy(&o, &hs, 2); return o;
}
static __device__ __forceinline__ float bf2f(bf16 b) {
  unsigned short hs; __builtin_memcpy(&hs, &b, 2);
  union { unsigned u; float f; } v; v.u = ((unsigned)hs) << 16; return v.f;
}
static __device__ __forceinline__ v8f zero8() {
  v8f z;
#pragma unroll
  for (int i = 0; i < 8; ++i) z[i] = 0.f;
  return z;
}
static __device__ __forceinline__ v16bf cat16(v8bf lo, v8bf hi) {
  return __builtin_shufflevector(lo, hi, 0,1,2,3,4,5,6,7,8,9,10,11,12,13,14,15);
}
// A fragment: 16 rows (M) x 32 (K), row-major source. Lane l<16 -> row l,
// K = kc + {0..7, 16..23}; lane l+16 -> row l, K = kc + {8..15, 24..31}.
static __device__ __forceinline__ v16bf load_a_frag(const bf16* base, int ld, int kc) {
  int lane = threadIdx.x & 31, l = lane & 15, hv = lane >> 4;
  const bf16* p = base + (size_t)l * ld + kc + hv * 8;
  return cat16(*(const v8bf*)p, *(const v8bf*)(p + 16));
}
// B fragment: source is Bt (N x K row-major, i.e. B transposed). Lane l<16 ->
// col l, K = kc + {0..7, 16..23}; lane l+16 -> col l, K = kc + {8..15, 24..31}.
static __device__ __forceinline__ v16bf load_b_frag(const bf16* base, int ld, int kc) {
  int lane = threadIdx.x & 31, l = lane & 15, hv = lane >> 4;
  const bf16* p = base + (size_t)l * ld + kc + hv * 8;
  return cat16(*(const v8bf*)p, *(const v8bf*)(p + 16));
}
static __device__ __forceinline__ v8f wmma_bf(v16bf a, v16bf b, v8f c) {
  return __builtin_amdgcn_wmma_f32_16x16x32_bf16(false, a, false, b, (short)0, c, false, false);
}

// ---- CDNA5 async global->LDS copy (ASYNCcnt path) ----
// LDS aperture generic pointers carry the LDS byte offset in addr[31:0].
static __device__ __forceinline__ void async_ld_b128(void* lds_ptr, const void* gptr) {
  unsigned off = (unsigned)(size_t)lds_ptr;
  asm volatile("global_load_async_to_lds_b128 %0, %1, off"
               :: "v"(off), "v"(gptr)
               : "memory");
}
static __device__ __forceinline__ void wait_async0() {
#if __has_builtin(__builtin_amdgcn_s_wait_asynccnt)
  __builtin_amdgcn_s_wait_asynccnt(0);
#else
  asm volatile("s_wait_asynccnt 0" ::: "memory");
#endif
}

// ---------------- conversion / transpose ----------------
__global__ void conv_f32_bf16(const float* __restrict__ src, bf16* __restrict__ dst, int n) {
  int i = blockIdx.x * blockDim.x + threadIdx.x;
  if (i < n) dst[i] = f2bf(src[i]);
}
// src: [R][C] f32 -> dst: [C][R] bf16
__global__ void transpose_bf16(const float* __restrict__ src, bf16* __restrict__ dst, int R, int C) {
  int i = blockIdx.x * blockDim.x + threadIdx.x;
  if (i >= R * C) return;
  int r = i / C, c = i % C;
  dst[(size_t)c * R + r] = f2bf(src[i]);
}

// ---------------- positional basis ----------------
__global__ void gamma_max_kernel(float* pmax) {
  int idx = blockIdx.x * blockDim.x + threadIdx.x;   // 2048*32
  float ad = (float)(idx >> 5);
  int i = idx & 31;
  float mean = 64.f * (i + 1);
  float cc = mean * 0.03125f;
  float conc = cc * cc;
  float rate = mean * (1.f / 1024.f);
  float logp = (conc - 1.f) * logf(ad) - rate * ad - (lgammaf(conc) - conc * logf(rate));
  float p = expf(logp) + 1e-8f;
#pragma unroll
  for (int off = 16; off >= 1; off >>= 1) p = fmaxf(p, __shfl_xor(p, off, 32));
  if ((threadIdx.x & 31) == 0) atomicMax((unsigned*)pmax, __float_as_uint(p));
}

__global__ void build_emb_kernel(bf16* __restrict__ emb, const float* __restrict__ pmax) {
  int idx = blockIdx.x * blockDim.x + threadIdx.x;   // 4096*96
  if (idx >= 4096 * 96) return;
  int j = idx / 96, f = idx % 96;
  float val = 0.f, val2 = 0.f;
  if (j < 4095) {
    float d = (float)(j - 2047);
    float ad = fabsf(d);
    int cls = f >> 5, i = f & 31;
    float v;
    if (cls == 0) {                       // exponential decay
      float hl = exp2f(3.0f + 8.0f * (float)i / 31.0f);
      v = exp2f(-ad / hl);
    } else if (cls == 1) {                // central mask
      float width = exp2f((float)(i + 1)) - 1.0f;
      v = (width > ad) ? 1.f : 0.f;
    } else {                              // gamma pdf
      float mean = 64.f * (i + 1);
      float cc = mean * 0.03125f;
      float conc = cc * cc;
      float rate = mean * (1.f / 1024.f);
      float logp = (conc - 1.f) * logf(ad) - rate * ad - (lgammaf(conc) - conc * logf(rate));
      v = (expf(logp) + 1e-8f) / pmax[0];
    }
    float s = (d > 0.f) ? 1.f : ((d < 0.f) ? -1.f : 0.f);
    val = v; val2 = s * v;
  }
  emb[(size_t)j * 192 + f]      = f2bf(val);
  emb[(size_t)j * 192 + 96 + f] = f2bf(val2);
}

// ---------------- generic WMMA GEMM ----------------
// Block: 8 waves share one 16-row A block (blockIdx.y); wave w owns cols
// (blockIdx.x*8 + w)*64 .. +63. A tile double-buffered in LDS via CDNA5
// async global->LDS copies (overlap stage s+1 copy with stage s WMMA).
// MODE 0: ->q bf16 (scaled)   1: ->k bf16   2: ->vT bf16   3: ->pos bf16   4: ->f32 out + bias
template<int MODE>
__global__ __launch_bounds__(256) void gemm_bf16(const bf16* __restrict__ A,
                                                 const bf16* __restrict__ Bt,
                                                 int K,
                                                 bf16* __restrict__ bout,
                                                 float* __restrict__ fout,
                                                 const float* __restrict__ bias) {
  __shared__ __attribute__((aligned(16))) bf16 Asmem[2][16 * 72];  // 72 = 64 + bank pad
  int wid = threadIdx.x >> 5;
  int lane = threadIdx.x & 31, l = lane & 15, hv = lane >> 4;
  int m0 = blockIdx.y << 4;
  int n0 = (blockIdx.x * 8 + wid) << 6;
  int S = K >> 6;                                     // stages of K=64

  v8f acc[4] = {zero8(), zero8(), zero8(), zero8()};

  // stage copier: threads 0..127 move 16 rows x 64 bf16 (one b128 each)
  int crow = threadIdx.x >> 3, ck = (threadIdx.x & 7) * 8;
  const bf16* Arow = A + (size_t)(m0 + crow) * K + ck;
  if (threadIdx.x < 128)
    async_ld_b128(&Asmem[0][crow * 72 + ck], Arow);

  for (int s = 0; s < S; ++s) {
    wait_async0();
    __syncthreads();
    if (s + 1 < S && threadIdx.x < 128)
      async_ld_b128(&Asmem[(s + 1) & 1][crow * 72 + ck], Arow + (s + 1) * 64);
    const bf16* ab = &Asmem[s & 1][0];
    int k0 = s * 64;
#pragma unroll
    for (int ks = 0; ks < 64; ks += 32) {
      v16bf a = load_a_frag(ab, 72, ks);
#pragma unroll
      for (int c = 0; c < 4; ++c) {
        const bf16* brow = Bt + (size_t)(n0 + 16 * c) * K;
        __builtin_prefetch(brow + (size_t)l * K + k0 + ks + 64, 0, 0);
        v16bf b = load_b_frag(brow, K, k0 + ks);
        acc[c] = wmma_bf(a, b, acc[c]);
      }
    }
  }

#pragma unroll
  for (int c = 0; c < 4; ++c) {
#pragma unroll
    for (int r = 0; r < 8; ++r) {
      float val = acc[c][r];
      int row = m0 + r + 8 * hv;
      int col = n0 + 16 * c + l;
      if (MODE == 0 || MODE == 1) {
        int b = row >> 11, nn = row & 2047, h = col >> 6, d = col & 63;
        float sc = (MODE == 0) ? 0.125f : 1.0f;   // DK^-0.5
        bout[(((size_t)(b * NHEAD + h) * NSEQ + nn) << 6) + d] = f2bf(val * sc);
      } else if (MODE == 2) {
        int b = row >> 11, nn = row & 2047, h = col >> 6, d = col & 63;
        bout[((size_t)(b * NHEAD + h) * HD + d) * NSEQ + nn] = f2bf(val);
      } else if (MODE == 3) {
        int h = col >> 6, d = col & 63;
        bout[(((size_t)h * NPOSP + row) << 6) + d] = f2bf(val);
      } else {
        fout[(size_t)row * NDIM + col] = val + bias[col];
      }
    }
  }
}

// ---------------- folded bias vectors ----------------
__global__ void ck_kernel(const bf16* __restrict__ kb, const float* __restrict__ cb,
                          float* __restrict__ ck) {
  int idx = blockIdx.x * blockDim.x + threadIdx.x;   // B*H*N
  if (idx >= 2 * NHEAD * NSEQ) return;
  int h = (idx >> 11) & 7;
  float s = 0.f;
#pragma unroll
  for (int d = 0; d < HD; ++d) s += bf2f(kb[(size_t)idx * HD + d]) * cb[h * HD + d];
  ck[idx] = s;
}
__global__ void pp_kernel(const bf16* __restrict__ pos, const float* __restrict__ pb,
                          float* __restrict__ pp) {
  int idx = blockIdx.x * blockDim.x + threadIdx.x;   // H*NPOSP
  if (idx >= NHEAD * NPOSP) return;
  int h = idx >> 12;
  float s = 0.f;
#pragma unroll
  for (int d = 0; d < HD; ++d) s += bf2f(pos[(size_t)idx * HD + d]) * pb[h * HD + d];
  pp[idx] = s;
}

// ---------------- flash attention ----------------
// Block = 8 waves, all sharing (b,h); wave w owns i-tile rows [ti0, ti0+16).
// K tile (32x64) and V tile (64x32) per j-step are block-shared: staged in
// LDS double-buffered via async global->LDS copies (8KB/stage, one pair of
// b128 copies per thread), overlapped with the 14 WMMAs of the prior stage.
__global__ __launch_bounds__(256) void flash_kernel(const bf16* __restrict__ q,
                                                    const bf16* __restrict__ k,
                                                    const bf16* __restrict__ vT,
                                                    const bf16* __restrict__ pos,
                                                    const float* __restrict__ ck,
                                                    const float* __restrict__ pp,
                                                    bf16* __restrict__ ob) {
  __shared__ float Rt[8][16][49];                              // banded rel scores
  __shared__ __attribute__((aligned(16))) bf16 Pt[8][16][40];  // P tile (C->A relayout)
  __shared__ __attribute__((aligned(16))) bf16 Ks[2][32][72];  // K stage buffers
  __shared__ __attribute__((aligned(16))) bf16 Vs[2][64][40];  // V stage buffers
  int wid = threadIdx.x >> 5;
  int lane = threadIdx.x & 31, l = lane & 15, hv = lane >> 4;
  int t = blockIdx.x * 8 + wid;                         // 0..2047
  int b = t >> 10;
  int h = (t >> 7) & 7;
  int ti0 = (t & 127) << 4;
  size_t bh = (size_t)(b * NHEAD + h);

  const bf16* qbase = q + (bh * NSEQ + ti0) * HD;
  v16bf aq0 = load_a_frag(qbase, HD, 0);
  v16bf aq1 = load_a_frag(qbase, HD, 32);

  const bf16* kbase0 = k + bh * NSEQ * HD;              // block-uniform
  const bf16* vbase0 = vT + bh * HD * NSEQ;             // block-uniform
  int krow = threadIdx.x >> 3, kk8 = (threadIdx.x & 7) * 8;   // 32 x 64 copier
  int vrow = threadIdx.x >> 2, vj8 = (threadIdx.x & 3) * 8;   // 64 x 32 copier

  float mrow[8], lrow[8];
  v8f o0 = zero8(), o1 = zero8(), o2 = zero8(), o3 = zero8();
#pragma unroll
  for (int r = 0; r < 8; ++r) { mrow[r] = -1e30f; lrow[r] = 0.f; }

  // prologue: stage 0
  async_ld_b128(&Ks[0][krow][kk8], kbase0 + (size_t)krow * HD + kk8);
  async_ld_b128(&Vs[0][vrow][vj8], vbase0 + (size_t)vrow * NSEQ + vj8);

  for (int s = 0; s < NSEQ / 32; ++s) {
    int j0 = s * 32;
    wait_async0();
    __syncthreads();
    if (s + 1 < NSEQ / 32) {
      int jn = j0 + 32;
      async_ld_b128(&Ks[(s + 1) & 1][krow][kk8],
                    kbase0 + (size_t)(jn + krow) * HD + kk8);
      async_ld_b128(&Vs[(s + 1) & 1][vrow][vj8],
                    vbase0 + (size_t)vrow * NSEQ + jn + vj8);
    }
    const bf16* kt = &Ks[s & 1][0][0];
    const bf16* vt = &Vs[s & 1][0][0];

    // ---- content logits: Q (16x64) . K^T -> 16x32 (K from LDS) ----
    v8f s0 = zero8(), s1 = zero8();
    s0 = wmma_bf(aq0, load_b_frag(kt, 72, 0), s0);
    s0 = wmma_bf(aq1, load_b_frag(kt, 72, 32), s0);
    s1 = wmma_bf(aq0, load_b_frag(kt + 16 * 72, 72, 0), s1);
    s1 = wmma_bf(aq1, load_b_frag(kt + 16 * 72, 72, 32), s1);

    // ---- banded rel logits: Q . pos[jp0 .. jp0+47]^T (u = tj + 15 - ti) ----
    int jp0 = (NSEQ - 1) + j0 - ti0 - 15;               // always in [0, 4048]
    const bf16* pbs = pos + ((size_t)h * NPOSP + jp0) * HD;
    v8f rt0 = zero8(), rt1 = zero8(), rt2 = zero8();
    rt0 = wmma_bf(aq0, load_b_frag(pbs, HD, 0), rt0);
    rt0 = wmma_bf(aq1, load_b_frag(pbs, HD, 32), rt0);
    rt1 = wmma_bf(aq0, load_b_frag(pbs + 16 * HD, HD, 0), rt1);
    rt1 = wmma_bf(aq1, load_b_frag(pbs + 16 * HD, HD, 32), rt1);
    rt2 = wmma_bf(aq0, load_b_frag(pbs + 32 * HD, HD, 0), rt2);
    rt2 = wmma_bf(aq1, load_b_frag(pbs + 32 * HD, HD, 32), rt2);

    // fold pp (per band column) and spill band to LDS for the shift
    float p0v = pp[h * NPOSP + jp0 + l];
    float p1v = pp[h * NPOSP + jp0 + 16 + l];
    float p2v = pp[h * NPOSP + jp0 + 32 + l];
#pragma unroll
    for (int r = 0; r < 8; ++r) {
      int m = r + 8 * hv;
      Rt[wid][m][l]      = rt0[r] + p0v;
      Rt[wid][m][16 + l] = rt1[r] + p1v;
      Rt[wid][m][32 + l] = rt2[r] + p2v;
    }
    float ck0 = ck[bh * NSEQ + j0 + l];
    float ck1 = ck[bh * NSEQ + j0 + 16 + l];

    float sf0[8], sf1[8];
#pragma unroll
    for (int r = 0; r < 8; ++r) {
      int m = r + 8 * hv;
      sf0[r] = s0[r] + ck0 + Rt[wid][m][l + 15 - m];        // relative_shift
      sf1[r] = s1[r] + ck1 + Rt[wid][m][16 + l + 15 - m];
    }

    // ---- online softmax (row = lane-half group of 16) ----
#pragma unroll
    for (int r = 0; r < 8; ++r) {
      float tmax = fmaxf(sf0[r], sf1[r]);
#pragma unroll
      for (int off = 8; off >= 1; off >>= 1) tmax = fmaxf(tmax, __shfl_xor(tmax, off, 16));
      float mnew = fmaxf(mrow[r], tmax);
      float corr = __expf(mrow[r] - mnew);
      float e0 = __expf(sf0[r] - mnew);
      float e1 = __expf(sf1[r] - mnew);
      float ps = e0 + e1;
#pragma unroll
      for (int off = 8; off >= 1; off >>= 1) ps += __shfl_xor(ps, off, 16);
      lrow[r] = lrow[r] * corr + ps;
      mrow[r] = mnew;
      o0[r] *= corr; o1[r] *= corr; o2[r] *= corr; o3[r] *= corr;
      int m = r + 8 * hv;
      Pt[wid][m][l]      = f2bf(e0);
      Pt[wid][m][16 + l] = f2bf(e1);
    }

    // ---- P (A-frag via LDS) . V^T (V from LDS) ----
    v16bf ap = load_a_frag(&Pt[wid][0][0], 40, 0);
    o0 = wmma_bf(ap, load_b_frag(vt,           40, 0), o0);
    o1 = wmma_bf(ap, load_b_frag(vt + 16 * 40, 40, 0), o1);
    o2 = wmma_bf(ap, load_b_frag(vt + 32 * 40, 40, 0), o2);
    o3 = wmma_bf(ap, load_b_frag(vt + 48 * 40, 40, 0), o3);
  }

#pragma unroll
  for (int r = 0; r < 8; ++r) {
    float inv = 1.f / lrow[r];
    int m = r + 8 * hv;
    size_t row = (size_t)b * NSEQ + ti0 + m;
    ob[row * HDK + h * HD + l]      = f2bf(o0[r] * inv);
    ob[row * HDK + h * HD + 16 + l] = f2bf(o1[r] * inv);
    ob[row * HDK + h * HD + 32 + l] = f2bf(o2[r] * inv);
    ob[row * HDK + h * HD + 48 + l] = f2bf(o3[r] * inv);
  }
}

// ---------------- launch ----------------
extern "C" void kernel_launch(void* const* d_in, const int* in_sizes, int n_in,
                              void* d_out, int out_size, void* d_ws, size_t ws_size,
                              hipStream_t stream) {
  (void)in_sizes; (void)n_in; (void)out_size; (void)ws_size;
  const float* x    = (const float*)d_in[0];
  const float* Wq   = (const float*)d_in[1];
  const float* Wk   = (const float*)d_in[2];
  const float* Wv   = (const float*)d_in[3];
  const float* Wo   = (const float*)d_in[4];
  const float* bo   = (const float*)d_in[5];
  const float* Wrel = (const float*)d_in[6];
  const float* cb   = (const float*)d_in[7];
  const float* pb   = (const float*)d_in[8];
  float* out = (float*)d_out;

  char* w = (char*)d_ws;
  auto alloc = [&](size_t bytes) { char* p = w; w += (bytes + 255) & ~(size_t)255; return p; };
  bf16* xb    = (bf16*)alloc((size_t)NTOK * NDIM * 2);
  bf16* WqT   = (bf16*)alloc((size_t)HDK * NDIM * 2);
  bf16* WkT   = (bf16*)alloc((size_t)HDK * NDIM * 2);
  bf16* WvT   = (bf16*)alloc((size_t)HDK * NDIM * 2);
  bf16* WrelT = (bf16*)alloc((size_t)HDK * 192 * 2);
  bf16* WoT   = (bf16*)alloc((size_t)NDIM * HDK * 2);
  bf16* qb    = (bf16*)alloc((size_t)NTOK * HDK * 2);
  bf16* kb    = (bf16*)alloc((size_t)NTOK * HDK * 2);
  bf16* vTb   = (bf16*)alloc((size_t)NTOK * HDK * 2);
  bf16* posb  = (bf16*)alloc((size_t)NHEAD * NPOSP * HD * 2);
  bf16* embb  = (bf16*)alloc((size_t)4096 * 192 * 2);
  bf16* oby   = (bf16*)alloc((size_t)NTOK * HDK * 2);
  float* ckv  = (float*)alloc((size_t)2 * NHEAD * NSEQ * 4);
  float* ppv  = (float*)alloc((size_t)NHEAD * NPOSP * 4);
  float* pmax = (float*)alloc(256);

  hipMemsetAsync(pmax, 0, 4, stream);

  conv_f32_bf16<<<(NTOK * NDIM + 255) / 256, 256, 0, stream>>>(x, xb, NTOK * NDIM);
  transpose_bf16<<<(NDIM * HDK + 255) / 256, 256, 0, stream>>>(Wq, WqT, NDIM, HDK);
  transpose_bf16<<<(NDIM * HDK + 255) / 256, 256, 0, stream>>>(Wk, WkT, NDIM, HDK);
  transpose_bf16<<<(NDIM * HDK + 255) / 256, 256, 0, stream>>>(Wv, WvT, NDIM, HDK);
  transpose_bf16<<<(192 * HDK + 255) / 256, 256, 0, stream>>>(Wrel, WrelT, 192, HDK);
  transpose_bf16<<<(HDK * NDIM + 255) / 256, 256, 0, stream>>>(Wo, WoT, HDK, NDIM);

  gamma_max_kernel<<<256, 256, 0, stream>>>(pmax);
  build_emb_kernel<<<(4096 * 96 + 255) / 256, 256, 0, stream>>>(embb, pmax);

  // QKV projections: grid (n-groups, m-tiles) = (512/512, 4096/16)
  gemm_bf16<0><<<dim3(1, 256), 256, 0, stream>>>(xb, WqT, NDIM, qb, nullptr, nullptr);
  gemm_bf16<1><<<dim3(1, 256), 256, 0, stream>>>(xb, WkT, NDIM, kb, nullptr, nullptr);
  gemm_bf16<2><<<dim3(1, 256), 256, 0, stream>>>(xb, WvT, NDIM, vTb, nullptr, nullptr);
  // pos = emb @ Wrel (M=4096, N=512, K=192)
  gemm_bf16<3><<<dim3(1, 256), 256, 0, stream>>>(embb, WrelT, 192, posb, nullptr, nullptr);

  ck_kernel<<<(2 * NHEAD * NSEQ + 255) / 256, 256, 0, stream>>>(kb, cb, ckv);
  pp_kernel<<<(NHEAD * NPOSP + 255) / 256, 256, 0, stream>>>(posb, pb, ppv);

  // flash attention: 2048 wave-tiles -> 256 blocks of 8 waves
  flash_kernel<<<256, 256, 0, stream>>>(qb, kb, vTb, posb, ckv, ppv, oby);

  // output projection (M=4096, N=1536, K=512): grid (1536/512, 256)
  gemm_bf16<4><<<dim3(3, 256), 256, 0, stream>>>(oby, WoT, HDK, nullptr, out, bo);
}
